// EncoderSS_47210280517647
// MI455X (gfx1250) — matrix operand order repaired and checked by reading, not compiled, not run on hardware
//
#include <hip/hip_runtime.h>
#include <math.h>

#define BATCH 8
#define SEQL0 512
#define DMODEL 512
#define LTOT 680
#define DFFN 2048
#define NHEAD 8
#define VTLD 704  // padded key stride for transposed V (32B-aligned f16 rows)

typedef _Float16 f16;
typedef __attribute__((ext_vector_type(16))) _Float16 v16h;
typedef __attribute__((ext_vector_type(8)))  _Float16 v8h;
typedef __attribute__((ext_vector_type(8)))  float    v8f;

// ================= f16 WMMA fragment loaders (wave32, 16x16x32) ============
// A (16x32): lanes0-15 row=lane&15; elems0-7 -> K=half*8+e, elems8-15 -> K=16+half*8+e
__device__ inline v16h ldA16(const f16* __restrict__ A, int lda,
                             int row_base, int k_base, int max_row) {
  int lane = threadIdx.x & 31;
  int half = lane >> 4;
  int r = row_base + (lane & 15);
  if (r > max_row) r = max_row;
  const f16* p = A + (size_t)r * lda + k_base + half * 8;
  v8h lo = *(const v8h*)p;
  v8h hi = *(const v8h*)(p + 16);
  return __builtin_shufflevector(lo, hi, 0, 1, 2, 3, 4, 5, 6, 7,
                                 8, 9, 10, 11, 12, 13, 14, 15);
}

// B (32x16) where B[k][n] = W[n][k], W row-major f16 [N,K]: lane -> col n, elems0-7 -> K=half*8+e, elems8-15 -> K=16+half*8+e
__device__ inline v16h ldB16(const f16* __restrict__ W, int ldw,
                             int n_base, int k_base, int max_n) {
  int lane = threadIdx.x & 31;
  int half = lane >> 4;
  int n = n_base + (lane & 15);
  if (n > max_n) n = max_n;
  const f16* p = W + (size_t)n * ldw + k_base + half * 8;
  v8h lo = *(const v8h*)p;
  v8h hi = *(const v8h*)(p + 16);
  return __builtin_shufflevector(lo, hi, 0, 1, 2, 3, 4, 5, 6, 7,
                                 8, 9, 10, 11, 12, 13, 14, 15);
}

// ================= generic f16 WMMA GEMM ====================================
// block = 256 threads = 8 waves; wave -> 16(M) x 64(N) via 4 accumulators.
// grid: x = N/64, y = ceil(M/128), z = batch.
template <int ACT, bool RESID, bool F16OUT, bool TOUT>
__global__ __launch_bounds__(256) void gemm16_kernel(
    const f16* __restrict__ A, int lda, size_t strideA,
    const f16* __restrict__ W, int ldw,
    const float* __restrict__ bias,
    const float* __restrict__ resid, int ldr, size_t strideR,
    void* __restrict__ Cv, int ldc, size_t strideC,
    int M, int N, int K) {
  int wave = threadIdx.x >> 5;
  int mt = blockIdx.y * 8 + wave;
  if (mt * 16 >= M) return;  // wave-uniform
  int nb = blockIdx.x * 64;
  int b = blockIdx.z;
  A += (size_t)b * strideA;
  if (RESID) resid += (size_t)b * strideR;

  v8f acc0 = {}, acc1 = {}, acc2 = {}, acc3 = {};
#pragma unroll 2
  for (int kk = 0; kk < K; kk += 32) {
    v16h a = ldA16(A, lda, mt * 16, kk, M - 1);
    v16h b0 = ldB16(W, ldw, nb, kk, N - 1);
    v16h b1 = ldB16(W, ldw, nb + 16, kk, N - 1);
    v16h b2 = ldB16(W, ldw, nb + 32, kk, N - 1);
    v16h b3 = ldB16(W, ldw, nb + 48, kk, N - 1);
    acc0 = __builtin_amdgcn_wmma_f32_16x16x32_f16(false, a, false, b0, (short)0, acc0, false, false);
    acc1 = __builtin_amdgcn_wmma_f32_16x16x32_f16(false, a, false, b1, (short)0, acc1, false, false);
    acc2 = __builtin_amdgcn_wmma_f32_16x16x32_f16(false, a, false, b2, (short)0, acc2, false, false);
    acc3 = __builtin_amdgcn_wmma_f32_16x16x32_f16(false, a, false, b3, (short)0, acc3, false, false);
  }

  int lane = threadIdx.x & 31, half = lane >> 4, l15 = lane & 15;
  float* Cf = (float*)Cv + (size_t)b * strideC;
  f16* Ch = (f16*)Cv + (size_t)b * strideC;
#pragma unroll
  for (int t = 0; t < 4; ++t) {
    v8f acc = (t == 0) ? acc0 : (t == 1) ? acc1 : (t == 2) ? acc2 : acc3;
    int col = nb + t * 16 + l15;
    float bv = (bias != nullptr && col < N) ? bias[col] : 0.0f;
#pragma unroll
    for (int j = 0; j < 8; ++j) {
      int row = mt * 16 + j + half * 8;
      if (row < M && col < N) {
        float val = acc[j] + bv;
        if (RESID) val += resid[(size_t)row * ldr + col];
        if (ACT == 1) val = 0.5f * val * (1.0f + erff(val * 0.70710678118f));
        size_t idx = TOUT ? ((size_t)col * ldc + row) : ((size_t)row * ldc + col);
        if (F16OUT) Ch[idx] = (f16)val;
        else        Cf[idx] = val;
      }
    }
  }
}

// ================= hierarchical mask (pure function of indices) ============
__device__ inline bool attn_allowed(int i, int j) {
  int li, pi, lj, pj;
  if (i < 512)      { li = 0; pi = i; }
  else if (i < 640) { li = 1; pi = i - 512; }
  else if (i < 672) { li = 2; pi = i - 640; }
  else              { li = 3; pi = i - 672; }
  if (j < 512)      { lj = 0; pj = j; }
  else if (j < 640) { lj = 1; pj = j - 512; }
  else if (j < 672) { lj = 2; pj = j - 640; }
  else              { lj = 3; pj = j - 672; }
  if (li == lj) { int d = pi - pj; return d <= 2 && d >= -2; }
  if (lj == li - 1) return (pj >> 2) == pi;
  if (li == lj - 1) return (pi >> 2) == pj;
  return false;
}

__device__ inline float rowmax16(float x) {
#pragma unroll
  for (int o = 8; o; o >>= 1) x = fmaxf(x, __shfl_xor(x, o, 16));
  return x;
}
__device__ inline float rowsum16(float x) {
#pragma unroll
  for (int o = 8; o; o >>= 1) x += __shfl_xor(x, o, 16);
  return x;
}

// ================= flash attention, all-f16 operands ========================
// q,k: f16 [B,680,512] (head-major cols); vt: f16 [B,512,704] transposed V.
__global__ __launch_bounds__(256) void attn_kernel(
    const f16* __restrict__ qg, const f16* __restrict__ kg,
    const f16* __restrict__ vt, f16* __restrict__ og) {
  __shared__ _Float16 plds[8][16][40];
  int wave = threadIdx.x >> 5;
  int qt = blockIdx.x * 8 + wave;
  if (qt * 16 >= LTOT) return;  // wave-uniform
  int h = blockIdx.y, b = blockIdx.z;
  int lane = threadIdx.x & 31, half = lane >> 4, l15 = lane & 15;

  const f16* Q = qg + ((size_t)b * LTOT) * DMODEL + h * 64;
  const f16* K = kg + ((size_t)b * LTOT) * DMODEL + h * 64;
  const f16* V = vt + (size_t)b * DMODEL * VTLD + (size_t)h * 64 * VTLD;

  v16h qa0 = ldA16(Q, DMODEL, qt * 16, 0, LTOT - 1);
  v16h qa1 = ldA16(Q, DMODEL, qt * 16, 32, LTOT - 1);

  v8f o0 = {}, o1 = {}, o2 = {}, o3 = {};
  float mrow[8], lrow[8];
#pragma unroll
  for (int j = 0; j < 8; ++j) { mrow[j] = -1e30f; lrow[j] = 0.0f; }

  for (int kt = 0; kt < 22; ++kt) {  // 22*32 = 704 >= 680 keys
    int kk0 = kt * 32;
    v8f s0 = {}, s1 = {};
    v16h kbf;
    kbf = ldB16(K, DMODEL, kk0, 0, LTOT - 1);
    s0 = __builtin_amdgcn_wmma_f32_16x16x32_f16(false, qa0, false, kbf, (short)0, s0, false, false);
    kbf = ldB16(K, DMODEL, kk0, 32, LTOT - 1);
    s0 = __builtin_amdgcn_wmma_f32_16x16x32_f16(false, qa1, false, kbf, (short)0, s0, false, false);
    kbf = ldB16(K, DMODEL, kk0 + 16, 0, LTOT - 1);
    s1 = __builtin_amdgcn_wmma_f32_16x16x32_f16(false, qa0, false, kbf, (short)0, s1, false, false);
    kbf = ldB16(K, DMODEL, kk0 + 16, 32, LTOT - 1);
    s1 = __builtin_amdgcn_wmma_f32_16x16x32_f16(false, qa1, false, kbf, (short)0, s1, false, false);

    int c0 = kk0 + l15, c1 = kk0 + 16 + l15;
#pragma unroll
    for (int j = 0; j < 8; ++j) {
      int row = qt * 16 + j + half * 8;
      int rq = row < LTOT ? row : LTOT - 1;
      float a0 = s0[j] * 0.125f;  // 1/sqrt(DK) folded here
      float a1 = s1[j] * 0.125f;
      if (c0 >= LTOT || !attn_allowed(rq, c0)) a0 = -1e9f;
      if (c1 >= LTOT || !attn_allowed(rq, c1)) a1 = -1e9f;
      float cmax = rowmax16(fmaxf(a0, a1));
      float mnew = fmaxf(mrow[j], cmax);
      float corr = expf(mrow[j] - mnew);
      float p0 = expf(a0 - mnew);
      float p1 = expf(a1 - mnew);
      lrow[j] = lrow[j] * corr + rowsum16(p0 + p1);
      mrow[j] = mnew;
      o0[j] *= corr; o1[j] *= corr; o2[j] *= corr; o3[j] *= corr;
      plds[wave][j + half * 8][l15] = (f16)p0;
      plds[wave][j + half * 8][16 + l15] = (f16)p1;
    }
    v16h pa;
#pragma unroll
    for (int e = 0; e < 8; ++e) pa[e] = plds[wave][l15][half * 8 + e];
#pragma unroll
    for (int e = 0; e < 8; ++e) pa[8 + e] = plds[wave][l15][16 + half * 8 + e];

    // V^T rows are channels; keys contiguous; pad keys 680..703 are zero.
    v16h vbf;
    vbf = ldB16(V, VTLD, 0, kk0, 511);
    o0 = __builtin_amdgcn_wmma_f32_16x16x32_f16(false, pa, false, vbf, (short)0, o0, false, false);
    vbf = ldB16(V, VTLD, 16, kk0, 511);
    o1 = __builtin_amdgcn_wmma_f32_16x16x32_f16(false, pa, false, vbf, (short)0, o1, false, false);
    vbf = ldB16(V, VTLD, 32, kk0, 511);
    o2 = __builtin_amdgcn_wmma_f32_16x16x32_f16(false, pa, false, vbf, (short)0, o2, false, false);
    vbf = ldB16(V, VTLD, 48, kk0, 511);
    o3 = __builtin_amdgcn_wmma_f32_16x16x32_f16(false, pa, false, vbf, (short)0, o3, false, false);
  }

  f16* O = og + ((size_t)b * LTOT) * DMODEL + h * 64;
#pragma unroll
  for (int j = 0; j < 8; ++j) {
    int row = qt * 16 + j + half * 8;
    if (row < LTOT) {
      float inv = 1.0f / lrow[j];
      O[(size_t)row * DMODEL + l15]      = (f16)(o0[j] * inv);
      O[(size_t)row * DMODEL + 16 + l15] = (f16)(o1[j] * inv);
      O[(size_t)row * DMODEL + 32 + l15] = (f16)(o2[j] * inv);
      O[(size_t)row * DMODEL + 48 + l15] = (f16)(o3[j] * inv);
    }
  }
}

// ================= elementwise f32 -> f16 conversion ========================
__global__ void cvt_kernel(const float* __restrict__ in, f16* __restrict__ out, int n) {
  int idx = blockIdx.x * blockDim.x + threadIdx.x;
  if (idx < n) out[idx] = (f16)in[idx];
}

// ================= embedding (writes f32 + f16) =============================
__global__ void embed_kernel(const float* __restrict__ seq,
                             const float* __restrict__ covW, const float* __restrict__ covb,
                             const float* __restrict__ convW, const float* __restrict__ convb,
                             float* __restrict__ x0, f16* __restrict__ x0h) {
  int idx = blockIdx.x * blockDim.x + threadIdx.x;
  if (idx >= BATCH * SEQL0 * DMODEL) return;
  int d = idx & 511;
  int l = (idx >> 9) & 511;
  int b = idx >> 18;
  const float* s = seq + ((size_t)b * SEQL0 + l) * 6;
  float c4 = s[5] / 370.0f - 0.5f;
  float acc = covb[d] + s[1] * covW[d * 5 + 0] + s[2] * covW[d * 5 + 1] +
              s[3] * covW[d * 5 + 2] + s[4] * covW[d * 5 + 3] + c4 * covW[d * 5 + 4];
  float dm1 = seq[((size_t)b * SEQL0 + ((l + 511) & 511)) * 6 + 0];
  float d00 = s[0];
  float dp1 = seq[((size_t)b * SEQL0 + ((l + 1) & 511)) * 6 + 0];
  acc += dm1 * convW[d * 3 + 0] + d00 * convW[d * 3 + 1] + dp1 * convW[d * 3 + 2] + convb[d];
  float pv = powf(10000.0f, 2.0f * (float)(d >> 1) / 512.0f);
  float ang = (float)l / pv;
  acc += (d & 1) ? cosf(ang) : sinf(ang);
  x0[idx] = acc;
  x0h[idx] = (f16)acc;
}

// ================= window-4 conv (tiny; VALU) ===============================
__global__ void conv_kernel(const float* __restrict__ in, const float* __restrict__ cw,
                            const float* __restrict__ cb, float* __restrict__ out,
                            int Li, int Lo) {
  int idx = blockIdx.x * blockDim.x + threadIdx.x;
  if (idx >= BATCH * 64 * Lo) return;
  int lg = idx % Lo;
  int o = (idx / Lo) & 63;
  int b = idx / (Lo * 64);
  const float* ip = in + (size_t)b * 64 * Li + lg * 4;
  const float* wp = cw + (size_t)o * 256;
  float acc = cb[o];
#pragma unroll 4
  for (int c = 0; c < 64; ++c) {
    const float* ic = ip + (size_t)c * Li;
    acc += ic[0] * wp[c * 4 + 0] + ic[1] * wp[c * 4 + 1] +
           ic[2] * wp[c * 4 + 2] + ic[3] * wp[c * 4 + 3];
  }
  out[idx] = acc;
}

// ================= batchnorm + ELU; writes transposed f16 copy ==============
__global__ void bn_elu_kernel(float* __restrict__ t, const float* __restrict__ g,
                              const float* __restrict__ bt, f16* __restrict__ tcat,
                              int Lo, int catOff) {
  __shared__ float sh1[256], sh2[256];
  int c = blockIdx.x;
  int n = BATCH * Lo;
  float s = 0.0f, sq = 0.0f;
  for (int i = threadIdx.x; i < n; i += 256) {
    int b = i / Lo, l = i % Lo;
    float v = t[((size_t)b * 64 + c) * Lo + l];
    s += v; sq += v * v;
  }
  sh1[threadIdx.x] = s; sh2[threadIdx.x] = sq;
  __syncthreads();
  for (int st = 128; st > 0; st >>= 1) {
    if (threadIdx.x < st) { sh1[threadIdx.x] += sh1[threadIdx.x + st]; sh2[threadIdx.x] += sh2[threadIdx.x + st]; }
    __syncthreads();
  }
  float mean = sh1[0] / (float)n;
  float var = sh2[0] / (float)n - mean * mean;
  float rstd = rsqrtf(var + 1e-5f);
  float gg = g[c], bb = bt[c];
  for (int i = threadIdx.x; i < n; i += 256) {
    int b = i / Lo, l = i % Lo;
    size_t p = ((size_t)b * 64 + c) * Lo + l;
    float v = (t[p] - mean) * rstd * gg + bb;
    v = v > 0.0f ? v : (expf(v) - 1.0f);  // ELU
    t[p] = v;
    tcat[((size_t)b * 168 + catOff + l) * 64 + c] = (f16)v;
  }
}

// ================= layernorm over D=512; writes f32 + f16 ===================
__global__ void ln_kernel(const float* __restrict__ in, const float* __restrict__ g,
                          const float* __restrict__ bt, float* __restrict__ out,
                          f16* __restrict__ out16, int rows, float eps) {
  __shared__ float sh1[256], sh2[256];
  int row = blockIdx.x, b = blockIdx.y;
  const float* p = in + ((size_t)b * rows + row) * DMODEL;
  size_t obase = ((size_t)b * rows + row) * DMODEL;
  float s = 0.0f, sq = 0.0f;
  for (int d = threadIdx.x; d < DMODEL; d += 256) { float v = p[d]; s += v; sq += v * v; }
  sh1[threadIdx.x] = s; sh2[threadIdx.x] = sq;
  __syncthreads();
  for (int st = 128; st > 0; st >>= 1) {
    if (threadIdx.x < st) { sh1[threadIdx.x] += sh1[threadIdx.x + st]; sh2[threadIdx.x] += sh2[threadIdx.x + st]; }
    __syncthreads();
  }
  float mean = sh1[0] / (float)DMODEL;
  float var = sh2[0] / (float)DMODEL - mean * mean;
  float rstd = rsqrtf(var + eps);
  for (int d = threadIdx.x; d < DMODEL; d += 256) {
    float v = (p[d] - mean) * rstd * g[d] + bt[d];
    out[obase + d] = v;
    out16[obase + d] = (f16)v;
  }
}

// ================= concat [x0; lo] + layernorm; writes f32 + f16 ============
__global__ void ln_concat_kernel(const float* __restrict__ x0, const float* __restrict__ lo,
                                 const float* __restrict__ g, const float* __restrict__ bt,
                                 float* __restrict__ out, f16* __restrict__ out16) {
  __shared__ float sh1[256], sh2[256];
  int row = blockIdx.x, b = blockIdx.y;
  const float* p = (row < 512) ? (x0 + ((size_t)b * 512 + row) * DMODEL)
                               : (lo + ((size_t)b * 168 + (row - 512)) * DMODEL);
  size_t obase = ((size_t)b * LTOT + row) * DMODEL;
  float s = 0.0f, sq = 0.0f;
  for (int d = threadIdx.x; d < DMODEL; d += 256) { float v = p[d]; s += v; sq += v * v; }
  sh1[threadIdx.x] = s; sh2[threadIdx.x] = sq;
  __syncthreads();
  for (int st = 128; st > 0; st >>= 1) {
    if (threadIdx.x < st) { sh1[threadIdx.x] += sh1[threadIdx.x + st]; sh2[threadIdx.x] += sh2[threadIdx.x + st]; }
    __syncthreads();
  }
  float mean = sh1[0] / (float)DMODEL;
  float var = sh2[0] / (float)DMODEL - mean * mean;
  float rstd = rsqrtf(var + 1e-5f);
  for (int d = threadIdx.x; d < DMODEL; d += 256) {
    float v = (p[d] - mean) * rstd * g[d] + bt[d];
    out[obase + d] = v;
    out16[obase + d] = (f16)v;
  }
}

// ================= output gather ============================================
__global__ void gather_kernel(const float* __restrict__ x, float* __restrict__ out) {
  int idx = blockIdx.x * blockDim.x + threadIdx.x;
  if (idx >= BATCH * SEQL0 * 4 * DMODEL) return;
  int d = idx & 511;
  int lvl = (idx >> 9) & 3;
  int i = (idx >> 11) & 511;
  int b = idx >> 20;
  int gi = (lvl == 0) ? i
         : (lvl == 1) ? 512 + (i >> 2)
         : (lvl == 2) ? 640 + (i >> 4)
                      : 672 + (i >> 6);
  out[idx] = x[((size_t)b * LTOT + gi) * DMODEL + d];
}

// ================= host orchestration =======================================
extern "C" void kernel_launch(void* const* d_in, const int* in_sizes, int n_in,
                              void* d_out, int out_size, void* d_ws, size_t ws_size,
                              hipStream_t stream) {
  const float* seq    = (const float*)d_in[0];
  const float* cov_W  = (const float*)d_in[1];
  const float* cov_b  = (const float*)d_in[2];
  const float* conv_W = (const float*)d_in[3];
  const float* conv_b = (const float*)d_in[4];
  const float* down_W = (const float*)d_in[5];
  const float* down_b = (const float*)d_in[6];
  const float* up_W   = (const float*)d_in[7];
  const float* up_b   = (const float*)d_in[8];
  const float* cw     = (const float*)d_in[9];
  const float* cb     = (const float*)d_in[10];
  const float* bn_g   = (const float*)d_in[11];
  const float* bn_b   = (const float*)d_in[12];
  const float* cn_g   = (const float*)d_in[13];
  const float* cn_b   = (const float*)d_in[14];
  const float* wq     = (const float*)d_in[15];
  const float* wk     = (const float*)d_in[16];
  const float* wv     = (const float*)d_in[17];
  const float* fcW    = (const float*)d_in[18];
  const float* fcb    = (const float*)d_in[19];
  const float* g1     = (const float*)d_in[20];
  const float* b1     = (const float*)d_in[21];
  const float* w1     = (const float*)d_in[22];
  const float* b1f    = (const float*)d_in[23];
  const float* w2     = (const float*)d_in[24];
  const float* b2f    = (const float*)d_in[25];
  const float* g2     = (const float*)d_in[26];
  const float* b2     = (const float*)d_in[27];

  char* base = (char*)d_ws;
  size_t off = 0;
  auto alloc = [&](size_t bytes) -> void* {
    void* p = base + off;
    off += (bytes + 255) & ~(size_t)255;
    return p;
  };
  // f32 buffers
  float* x0  = (float*)alloc((size_t)BATCH * 512 * 512 * 4);
  float* t0  = (float*)alloc((size_t)BATCH * 64 * 512 * 4);
  float* t1  = (float*)alloc((size_t)BATCH * 64 * 128 * 4);
  float* t2  = (float*)alloc((size_t)BATCH * 64 * 32 * 4);
  float* t3  = (float*)alloc((size_t)BATCH * 64 * 8 * 4);
  float* lo  = (float*)alloc((size_t)BATCH * 168 * 512 * 4);
  float* x   = (float*)alloc((size_t)BATCH * LTOT * 512 * 4);
  float* tmp = (float*)alloc((size_t)BATCH * LTOT * 512 * 4);
  // f16 buffers
  f16* x0h    = (f16*)alloc((size_t)BATCH * 512 * 512 * 2);
  f16* tcat16 = (f16*)alloc((size_t)BATCH * 168 * 64 * 2);
  f16* xh     = (f16*)alloc((size_t)BATCH * LTOT * 512 * 2);
  f16* qh     = (f16*)alloc((size_t)BATCH * LTOT * 512 * 2);
  f16* kh     = (f16*)alloc((size_t)BATCH * LTOT * 512 * 2);
  f16* vt     = (f16*)alloc((size_t)BATCH * DMODEL * VTLD * 2);
  f16* oh     = (f16*)alloc((size_t)BATCH * LTOT * 512 * 2);
  f16* hh     = (f16*)alloc((size_t)BATCH * LTOT * DFFN * 2);
  // f16 weights (converted once per launch)
  f16* downWh = (f16*)alloc((size_t)64 * 512 * 2);
  f16* upWh   = (f16*)alloc((size_t)512 * 64 * 2);
  f16* wqh    = (f16*)alloc((size_t)4 * 512 * 512 * 2);
  f16* wkh    = (f16*)alloc((size_t)4 * 512 * 512 * 2);
  f16* wvh    = (f16*)alloc((size_t)4 * 512 * 512 * 2);
  f16* fcWh   = (f16*)alloc((size_t)4 * 512 * 512 * 2);
  f16* w1h    = (f16*)alloc((size_t)4 * DFFN * 512 * 2);
  f16* w2h    = (f16*)alloc((size_t)4 * 512 * DFFN * 2);

  auto cvt = [&](const float* in, f16* out, size_t n) {
    cvt_kernel<<<(unsigned)((n + 255) / 256), 256, 0, stream>>>(in, out, (int)n);
  };
  cvt(down_W, downWh, (size_t)64 * 512);
  cvt(up_W, upWh, (size_t)512 * 64);
  cvt(wq, wqh, (size_t)4 * 512 * 512);
  cvt(wk, wkh, (size_t)4 * 512 * 512);
  cvt(wv, wvh, (size_t)4 * 512 * 512);
  cvt(fcW, fcWh, (size_t)4 * 512 * 512);
  cvt(w1, w1h, (size_t)4 * DFFN * 512);
  cvt(w2, w2h, (size_t)4 * 512 * DFFN);

  // zero-fill padded transposed-V once (pad keys 680..703 stay zero forever)
  hipMemsetAsync(vt, 0, (size_t)BATCH * DMODEL * VTLD * 2, stream);

  // 1) embedding
  embed_kernel<<<(BATCH * 512 * 512 + 255) / 256, 256, 0, stream>>>(
      seq, cov_W, cov_b, conv_W, conv_b, x0, x0h);

  // 2) down-projection -> transposed f32 (B,64,512): M=512,N=64,K=512
  gemm16_kernel<0, false, false, true><<<dim3(1, 4, BATCH), 256, 0, stream>>>(
      x0h, 512, (size_t)512 * 512, downWh, 512, down_b,
      nullptr, 0, 0, t0, 512, (size_t)64 * 512, 512, 64, 512);

  // 3) conv/BN/ELU pyramid
  conv_kernel<<<(BATCH * 64 * 128 + 255) / 256, 256, 0, stream>>>(t0, cw, cb, t1, 512, 128);
  bn_elu_kernel<<<64, 256, 0, stream>>>(t1, bn_g, bn_b, tcat16, 128, 0);
  conv_kernel<<<(BATCH * 64 * 32 + 255) / 256, 256, 0, stream>>>(t1, cw + 16384, cb + 64, t2, 128, 32);
  bn_elu_kernel<<<64, 256, 0, stream>>>(t2, bn_g + 64, bn_b + 64, tcat16, 32, 128);
  conv_kernel<<<(BATCH * 64 * 8 + 255) / 256, 256, 0, stream>>>(t2, cw + 32768, cb + 128, t3, 32, 8);
  bn_elu_kernel<<<64, 256, 0, stream>>>(t3, bn_g + 128, bn_b + 128, tcat16, 8, 160);

  // 4) up-projection: M=168,N=512,K=64 -> lo f32
  gemm16_kernel<0, false, false, false><<<dim3(8, 2, BATCH), 256, 0, stream>>>(
      tcat16, 64, (size_t)168 * 64, upWh, 64, up_b,
      nullptr, 0, 0, lo, 512, (size_t)168 * 512, 168, 512, 64);

  // 5) concat + layernorm -> x (f32) + xh (f16)
  ln_concat_kernel<<<dim3(LTOT, BATCH), 256, 0, stream>>>(x0, lo, cn_g, cn_b, x, xh);

  // 6) transformer layers
  for (int i = 0; i < 4; ++i) {
    size_t wo = (size_t)i * 512 * 512;
    // Q,K f16 out: M=680,N=512,K=512
    gemm16_kernel<0, false, true, false><<<dim3(8, 6, BATCH), 256, 0, stream>>>(
        xh, 512, (size_t)LTOT * 512, wqh + wo, 512, nullptr,
        nullptr, 0, 0, qh, 512, (size_t)LTOT * 512, LTOT, 512, 512);
    gemm16_kernel<0, false, true, false><<<dim3(8, 6, BATCH), 256, 0, stream>>>(
        xh, 512, (size_t)LTOT * 512, wkh + wo, 512, nullptr,
        nullptr, 0, 0, kh, 512, (size_t)LTOT * 512, LTOT, 512, 512);
    // V f16, written transposed [channel][key] with padded stride 704
    gemm16_kernel<0, false, true, true><<<dim3(8, 6, BATCH), 256, 0, stream>>>(
        xh, 512, (size_t)LTOT * 512, wvh + wo, 512, nullptr,
        nullptr, 0, 0, vt, VTLD, (size_t)DMODEL * VTLD, LTOT, 512, 512);
    // masked flash attention -> oh f16
    attn_kernel<<<dim3(6, NHEAD, BATCH), 256, 0, stream>>>(qh, kh, vt, oh);
    // FC + residual -> tmp f32, then LN -> x + xh
    gemm16_kernel<0, true, false, false><<<dim3(8, 6, BATCH), 256, 0, stream>>>(
        oh, 512, (size_t)LTOT * 512, fcWh + wo, 512, fcb + i * 512,
        x, 512, (size_t)LTOT * 512, tmp, 512, (size_t)LTOT * 512, LTOT, 512, 512);
    ln_kernel<<<dim3(LTOT, BATCH), 256, 0, stream>>>(tmp, g1 + i * 512, b1 + i * 512, x, xh, LTOT, 1e-6f);
    // FFN: GELU(x@w1^T+b1f) -> hh f16 ; hh@w2^T+b2f + x -> tmp ; LN -> x + xh
    gemm16_kernel<1, false, true, false><<<dim3(32, 6, BATCH), 256, 0, stream>>>(
        xh, 512, (size_t)LTOT * 512, w1h + (size_t)i * DFFN * 512, 512, b1f + i * DFFN,
        nullptr, 0, 0, hh, DFFN, (size_t)LTOT * DFFN, LTOT, DFFN, 512);
    gemm16_kernel<0, true, false, false><<<dim3(8, 6, BATCH), 256, 0, stream>>>(
        hh, DFFN, (size_t)LTOT * DFFN, w2h + (size_t)i * 512 * DFFN, DFFN, b2f + i * 512,
        x, 512, (size_t)LTOT * 512, tmp, 512, (size_t)LTOT * 512, LTOT, 512, DFFN);
    ln_kernel<<<dim3(LTOT, BATCH), 256, 0, stream>>>(tmp, g2 + i * 512, b2 + i * 512, x, xh, LTOT, 1e-6f);
  }

  // 7) output gather (B, 512, 4*512)
  gather_kernel<<<(BATCH * SEQL0 * 4 * DMODEL + 255) / 256, 256, 0, stream>>>(x, (float*)d_out);
}
